// GCNEncoder_4827543241243
// MI455X (gfx1250) — matrix operand order repaired and checked by reading, not compiled, not run on hardware
//
#include <hip/hip_runtime.h>

// ---------- types ----------
typedef __attribute__((ext_vector_type(16))) __bf16 v16b;
typedef __attribute__((ext_vector_type(8)))  __bf16 v8b;
typedef __attribute__((ext_vector_type(8)))  float  v8f;
typedef __attribute__((ext_vector_type(4)))  unsigned int v4u;
typedef __attribute__((ext_vector_type(8)))  int v8i;
typedef __attribute__((ext_vector_type(4)))  int v4i;

#define IN_DIM     256
#define HIDDEN_DIM 128
#define LATENT_DIM 64

// ---------- Tensor Data Mover: 1-row tile DMA global -> LDS ----------
// D# per CDNA5 ISA:
//  group0: [1:0]=count=1, [63:32]=lds_addr, [120:64]=global_addr, [127:126]=type=2
//  group1: data_size=3 (8B units), tensor_dim0 = tile_dim0 = n8, tensor_dim1 = tile_dim1 = 1,
//          tensor_dim0_stride = n8
__device__ inline void tdm_load_to_lds(unsigned lds_off, const void* gptr, unsigned n8) {
    unsigned long long ga = (unsigned long long)gptr;
    v4u g0;
    g0[0] = 1u;                                              // count=1
    g0[1] = lds_off;                                         // lds_addr
    g0[2] = (unsigned)(ga & 0xffffffffu);                    // global_addr[31:0]
    g0[3] = (unsigned)((ga >> 32) & 0x1ffffffu) | (2u << 30);// global_addr[56:32] | type=2
    v8i g1;
    g1[0] = (3 << 16);                                       // data_size = 8 bytes
    g1[1] = (int)((n8 & 0xffffu) << 16);                     // tensor_dim0[15:0]
    g1[2] = (int)((n8 >> 16) & 0xffffu) | (1 << 16);         // tensor_dim0[31:16], tensor_dim1=1
    g1[3] = (int)((n8 & 0xffffu) << 16);                     // tile_dim0 = n8
    g1[4] = 1;                                               // tile_dim1 = 1
    g1[5] = (int)n8;                                         // tensor_dim0_stride[31:0]
    g1[6] = 0;
    g1[7] = 0;
    v4i z4 = {0, 0, 0, 0};
    v8i z8 = {0, 0, 0, 0, 0, 0, 0, 0};
    __builtin_amdgcn_tensor_load_to_lds(g0, g1, z4, z4, z8, 0);   // 6-arg toolchain variant
}

// ---------- small utility kernels ----------
__global__ __launch_bounds__(256) void fill_zero_kernel(float* __restrict__ p, int n) {
    int i = blockIdx.x * 256 + threadIdx.x;
    if (i < n) p[i] = 0.0f;
}

__global__ __launch_bounds__(256) void degree_kernel(const long long* __restrict__ dst,
                                                     float* __restrict__ deg, int nedges) {
    int e = blockIdx.x * 256 + threadIdx.x;
    if (e < nedges) atomicAdd(&deg[(int)dst[e]], 1.0f);
}

__global__ __launch_bounds__(256) void rsqrt_kernel(float* __restrict__ dinv, int n) {
    int i = blockIdx.x * 256 + threadIdx.x;
    if (i < n) dinv[i] = rsqrtf(dinv[i] + 1.0f);   // +1 self loop
}

// Pack a row-major fp32 weight [KDIM x NDIM] into bf16 WMMA B-fragment order:
// packed[((kt*NT + nt)*32 + lane)*16 + e]  where for lane l, element e:
//   K = kt*32 + (e/8)*16 + (l/16)*8 + (e%8) ,  N = nt*16 + (l%16)
template <int KDIM, int NDIM>
__global__ __launch_bounds__(256) void pack_w_kernel(const float* __restrict__ W,
                                                     __bf16* __restrict__ out) {
    constexpr int NT = NDIM / 16;
    int p = blockIdx.x * 256 + threadIdx.x;
    if (p >= KDIM * NDIM) return;
    int e    = p & 15;
    int lane = (p >> 4) & 31;
    int t    = p >> 9;                 // (kt*NT + nt)
    int nt   = t % NT;
    int kt   = t / NT;
    int K = kt * 32 + (e >> 3) * 16 + (lane >> 4) * 8 + (e & 7);
    int N = nt * 16 + (lane & 15);
    out[p] = (__bf16)W[K * NDIM + N];
}

// ---------- A-fragment loaders (16x32 bf16 layout, ISA 7.12.2) ----------
// lane l (group g = l/16) needs K runs [off, off+8) and [off+16, off+24)
// with off = kk + g*8, relative to its row.
__device__ inline v16b load_a_frag(const float* __restrict__ p) {
    const float4 a0 = *(const float4*)(p);
    const float4 a1 = *(const float4*)(p + 4);
    const float4 a2 = *(const float4*)(p + 16);
    const float4 a3 = *(const float4*)(p + 20);
    v16b r;
    r[0]  = (__bf16)a0.x; r[1]  = (__bf16)a0.y; r[2]  = (__bf16)a0.z; r[3]  = (__bf16)a0.w;
    r[4]  = (__bf16)a1.x; r[5]  = (__bf16)a1.y; r[6]  = (__bf16)a1.z; r[7]  = (__bf16)a1.w;
    r[8]  = (__bf16)a2.x; r[9]  = (__bf16)a2.y; r[10] = (__bf16)a2.z; r[11] = (__bf16)a2.w;
    r[12] = (__bf16)a3.x; r[13] = (__bf16)a3.y; r[14] = (__bf16)a3.z; r[15] = (__bf16)a3.w;
    return r;
}

__device__ inline v16b load_a_frag(const __bf16* __restrict__ p) {
    v8b lo = *(const v8b*)(p);
    v8b hi = *(const v8b*)(p + 16);
    v16b r;
#pragma unroll
    for (int i = 0; i < 8; ++i) { r[i] = lo[i]; r[8 + i] = hi[i]; }
    return r;
}

// ---------- WMMA GEMM: out[M x NDIM] = A[M x KDIM] @ W  (bf16 x bf16 -> f32) ----------
// Block = 256 threads = 8 waves; packed W is DMA'd to LDS once per block via the
// Tensor Data Mover, then every wave feeds its B fragments from ds_load_b128.
template <int KDIM, int NDIM, typename AT>
__global__ __launch_bounds__(256) void gemm_wmma_kernel(const AT* __restrict__ A,
                                                        const __bf16* __restrict__ Wp,
                                                        float* __restrict__ out, int nrows) {
    constexpr int NT = NDIM / 16;
    constexpr int KT = KDIM / 32;
    __shared__ __align__(32) __bf16 lds_w[KDIM * NDIM];

    // one TDM issue per workgroup (EXEC-ignored op; wave 0 waits on TENSORcnt)
    if (threadIdx.x == 0) {
        tdm_load_to_lds((unsigned)(unsigned long long)&lds_w[0], (const void*)Wp,
                        (unsigned)(KDIM * NDIM * 2 / 8));
        __builtin_amdgcn_s_wait_tensorcnt(0);
    }
    __syncthreads();

    const int lane = threadIdx.x & 31;
    const int wave = threadIdx.x >> 5;
    const int m    = lane & 15;
    const int grp  = lane >> 4;

    const int row0 = blockIdx.x * 128 + wave * 16;
    int arow = row0 + m;
    if (arow >= nrows) arow = nrows - 1;              // clamp: keep EXEC all-ones
    const AT* aptr = A + (size_t)arow * KDIM;

    v8f zero;
#pragma unroll
    for (int i = 0; i < 8; ++i) zero[i] = 0.0f;
    v8f acc[NT];
#pragma unroll
    for (int nt = 0; nt < NT; ++nt) acc[nt] = zero;

#pragma unroll
    for (int kt = 0; kt < KT; ++kt) {
        const int off = kt * 32 + grp * 8;
        if (kt + 1 < KT) __builtin_prefetch(aptr + off + 32, 0, 0);
        v16b af = load_a_frag(aptr + off);
        v16b bfr[NT];
#pragma unroll
        for (int nt = 0; nt < NT; ++nt)
            bfr[nt] = *(const v16b*)(lds_w + ((kt * NT + nt) * 32 + lane) * 16);
#pragma unroll
        for (int nt = 0; nt < NT; ++nt)
            acc[nt] = __builtin_amdgcn_wmma_f32_16x16x32_bf16(
                false, af, false, bfr[nt], (short)0, acc[nt], false, false);
    }

    // C/D layout: VGPR r -> M = grp*8 + r, N = m
#pragma unroll
    for (int nt = 0; nt < NT; ++nt) {
#pragma unroll
        for (int r = 0; r < 8; ++r) {
            int orow = row0 + grp * 8 + r;
            if (orow < nrows) out[(size_t)orow * NDIM + nt * 16 + m] = acc[nt][r];
        }
    }
}

// ---------- edge aggregation: agg[dst] += h[src] * dinv[src]*dinv[dst] ----------
template <int DIM>
__global__ __launch_bounds__(256) void edge_agg_kernel(const float* __restrict__ h,
                                                       const long long* __restrict__ src,
                                                       const long long* __restrict__ dst,
                                                       const float* __restrict__ dinv,
                                                       float* __restrict__ agg, int nedges) {
    constexpr int TPE = DIM / 4;                       // threads per edge (float4 each)
    int t = blockIdx.x * 256 + threadIdx.x;
    int e = t / TPE;
    if (e >= nedges) return;
    int f = (t % TPE) * 4;
    int s = (int)src[e];
    int d = (int)dst[e];
    float nrm = dinv[s] * dinv[d];
    const float4 v = *(const float4*)(h + (size_t)s * DIM + f);
    float* ap = agg + (size_t)d * DIM + f;
    atomicAdd(ap + 0, v.x * nrm);
    atomicAdd(ap + 1, v.y * nrm);
    atomicAdd(ap + 2, v.z * nrm);
    atomicAdd(ap + 3, v.w * nrm);
}

// ---------- layer epilogues ----------
// layer 1: h_bf16 = relu(agg1 + h1 * dinv^2 + b1)
__global__ __launch_bounds__(256) void post1_kernel(const float* __restrict__ agg,
                                                    const float* __restrict__ h,
                                                    const float* __restrict__ dinv,
                                                    const float* __restrict__ b,
                                                    __bf16* __restrict__ hb, int total) {
    int i = blockIdx.x * 256 + threadIdx.x;
    if (i >= total) return;
    int node = i >> 7;                 // / 128
    int c    = i & 127;
    float dd = dinv[node];
    float v  = agg[i] + h[i] * dd * dd + b[c];
    hb[i] = (__bf16)fmaxf(v, 0.0f);
}

// layer 2: out = agg2 + h2 * dinv^2 + b2
__global__ __launch_bounds__(256) void post2_kernel(const float* __restrict__ agg,
                                                    const float* __restrict__ h,
                                                    const float* __restrict__ dinv,
                                                    const float* __restrict__ b,
                                                    float* __restrict__ out, int total) {
    int i = blockIdx.x * 256 + threadIdx.x;
    if (i >= total) return;
    int node = i >> 6;                 // / 64
    int c    = i & 63;
    float dd = dinv[node];
    out[i] = agg[i] + h[i] * dd * dd + b[c];
}

// ---------- host launcher ----------
static inline size_t align_up(size_t x) { return (x + 255) & ~(size_t)255; }
static inline int cdiv(long long a, long long b) { return (int)((a + b - 1) / b); }

extern "C" void kernel_launch(void* const* d_in, const int* in_sizes, int n_in,
                              void* d_out, int out_size, void* d_ws, size_t ws_size,
                              hipStream_t stream) {
    const float*     x    = (const float*)d_in[0];
    const long long* ei   = (const long long*)d_in[1];   // int64 per reference
    const float*     W1   = (const float*)d_in[2];
    const float*     b1   = (const float*)d_in[3];
    const float*     W2   = (const float*)d_in[4];
    const float*     b2   = (const float*)d_in[5];
    float*           out  = (float*)d_out;

    const int nn = in_sizes[0] / IN_DIM;   // 100000
    const int ne = in_sizes[1] / 2;        // 1600000
    const long long* srcp = ei;
    const long long* dstp = ei + ne;

    // workspace carve-out
    char* ws = (char*)d_ws;
    size_t o = 0;
    float*  dinv = (float*)(ws + o);  o += align_up((size_t)nn * 4);
    __bf16* W1p  = (__bf16*)(ws + o); o += align_up((size_t)IN_DIM * HIDDEN_DIM * 2);
    __bf16* W2p  = (__bf16*)(ws + o); o += align_up((size_t)HIDDEN_DIM * LATENT_DIM * 2);
    float*  h1   = (float*)(ws + o);  o += align_up((size_t)nn * HIDDEN_DIM * 4);
    float*  agg1 = (float*)(ws + o);  o += align_up((size_t)nn * HIDDEN_DIM * 4);
    __bf16* hb   = (__bf16*)(ws + o); o += align_up((size_t)nn * HIDDEN_DIM * 2);
    float*  h2   = (float*)(ws + o);  o += align_up((size_t)nn * LATENT_DIM * 4);
    float*  agg2 = (float*)(ws + o);  o += align_up((size_t)nn * LATENT_DIM * 4);
    (void)ws_size; (void)n_in; (void)out_size;

    // 1) zero deg and aggregation buffers (harness does not re-zero between replays)
    fill_zero_kernel<<<cdiv(nn, 256), 256, 0, stream>>>(dinv, nn);
    fill_zero_kernel<<<cdiv((long long)nn * HIDDEN_DIM, 256), 256, 0, stream>>>(agg1, nn * HIDDEN_DIM);
    fill_zero_kernel<<<cdiv((long long)nn * LATENT_DIM, 256), 256, 0, stream>>>(agg2, nn * LATENT_DIM);

    // 2) pack weights into bf16 WMMA B-fragment order
    pack_w_kernel<IN_DIM, HIDDEN_DIM><<<cdiv(IN_DIM * HIDDEN_DIM, 256), 256, 0, stream>>>(W1, W1p);
    pack_w_kernel<HIDDEN_DIM, LATENT_DIM><<<cdiv(HIDDEN_DIM * LATENT_DIM, 256), 256, 0, stream>>>(W2, W2p);

    // 3) degrees -> dinv = rsqrt(deg + 1)
    degree_kernel<<<cdiv(ne, 256), 256, 0, stream>>>(dstp, dinv, ne);
    rsqrt_kernel<<<cdiv(nn, 256), 256, 0, stream>>>(dinv, nn);

    // 4) layer 1: h1 = x @ W1 (WMMA bf16 -> f32, W staged via TDM)
    gemm_wmma_kernel<IN_DIM, HIDDEN_DIM, float>
        <<<cdiv(nn, 128), 256, 0, stream>>>(x, W1p, h1, nn);

    // 5) layer-1 neighbor aggregation (L2-resident scatter-add)
    edge_agg_kernel<HIDDEN_DIM>
        <<<cdiv((long long)ne * (HIDDEN_DIM / 4), 256), 256, 0, stream>>>(h1, srcp, dstp, dinv, agg1, ne);

    // 6) self-loop + bias + relu, emit bf16 activations for GEMM2
    post1_kernel<<<cdiv((long long)nn * HIDDEN_DIM, 256), 256, 0, stream>>>(
        agg1, h1, dinv, b1, hb, nn * HIDDEN_DIM);

    // 7) layer 2: h2 = h @ W2 (WMMA bf16 -> f32, W staged via TDM)
    gemm_wmma_kernel<HIDDEN_DIM, LATENT_DIM, __bf16>
        <<<cdiv(nn, 128), 256, 0, stream>>>(hb, W2p, h2, nn);

    // 8) layer-2 neighbor aggregation
    edge_agg_kernel<LATENT_DIM>
        <<<cdiv((long long)ne * (LATENT_DIM / 4), 256), 256, 0, stream>>>(h2, srcp, dstp, dinv, agg2, ne);

    // 9) final self-loop + bias -> d_out
    post2_kernel<<<cdiv((long long)nn * LATENT_DIM, 256), 256, 0, stream>>>(
        agg2, h2, dinv, b2, out, nn * LATENT_DIM);
}
